// CircuitAwareAttention_6021544149160
// MI455X (gfx1250) — matrix operand order repaired and checked by reading, not compiled, not run on hardware
//
#include <hip/hip_runtime.h>
#include <hip/hip_bf16.h>

// ---------------------------------------------------------------------------
// CircuitAwareAttention for MI455X (gfx1250, wave32, WMMA 16x16x32 bf16)
//
// Pipeline:
//   1) prep_x       : xb = bf16(x + pos_enc)                       [4096,512]
//   2) gemm_kernel  : q/k/v = xb @ W.T + b  (bf16, head-split)     x3
//                     A-tile staged to LDS via TENSOR_LOAD_TO_LDS (TDM)
//   3) transpose_v  : vt[b,h,dh,n] = v[b,h,n,dh]
//   4) attn_kernel  : flash attention + topology bias + mask       (WMMA)
//   5) gemm_kernel  : out = attn @ Wo.T + bo  (f32 -> d_out)
// ---------------------------------------------------------------------------

typedef __bf16 bf16_t;
typedef __attribute__((ext_vector_type(16))) __bf16 v16bf;
typedef __attribute__((ext_vector_type(8)))  __bf16 v8bf;
typedef __attribute__((ext_vector_type(8)))  float  v8f;
typedef __attribute__((ext_vector_type(4)))  unsigned u32x4;
typedef __attribute__((ext_vector_type(8)))  unsigned u32x8;

union Frag {
    v16bf v;
    v8bf  h[2];
};

#define N_NODES  2048
#define D_MODEL  512
#define N_HEADS  8
#define HEAD_DIM 64
#define M_TOTAL  4096   // BATCH * N_NODES

// ---------------------------------------------------------------------------
// TDM: DMA a [rows x 32] bf16 tile (row stride 512 elems) from global -> LDS.
// Issued by one wave; caller must s_wait_tensorcnt + workgroup barrier.
// D# layout per CDNA5 ISA ch.8 (group0: 4 SGPRs, group1: 8 SGPRs, 2D tensor).
// ---------------------------------------------------------------------------
__device__ __forceinline__ void tdm_load_tile_bf16(
    const bf16_t* gptr, unsigned lds_off)
{
    unsigned long long ga = (unsigned long long)(uintptr_t)gptr;

    u32x4 g0;
    g0[0] = 1u;                                        // count=1 (valid), user D#
    g0[1] = lds_off;                                   // lds_addr (bytes)
    g0[2] = (unsigned)(ga & 0xffffffffu);              // global_addr[31:0]
    g0[3] = (unsigned)((ga >> 32) & 0x01ffffffu)       // global_addr[56:32]
          | (2u << 30);                                // type = 2 ("image")

    u32x8 g1;
    g1[0] = 1u << 16;                                  // data_size = 1 (2 bytes)
    g1[1] = (unsigned)(D_MODEL & 0xffff) << 16;        // tensor_dim0[15:0] = 512
    g1[2] = (unsigned)(D_MODEL >> 16)                  // tensor_dim0[31:16]
          | ((unsigned)(M_TOTAL & 0xffff) << 16);      // tensor_dim1[15:0] = 4096
    g1[3] = (unsigned)(M_TOTAL >> 16)                  // tensor_dim1[31:16]
          | (32u << 16);                               // tile_dim0 = 32
    g1[4] = 128u;                                      // tile_dim1 = 128 rows
    g1[5] = (unsigned)D_MODEL;                         // tensor_dim0_stride[31:0]
    g1[6] = 0u;                                        // stride hi / dim1_stride lo
    g1[7] = 0u;                                        // dim1_stride hi

    asm volatile("tensor_load_to_lds %0, %1" :: "s"(g0), "s"(g1) : "memory");
}

// ---------------------------------------------------------------------------
// 1) xb = bf16(x + pos_enc)
// ---------------------------------------------------------------------------
__global__ __launch_bounds__(256) void prep_x_kernel(
    const float* __restrict__ x, const float* __restrict__ pos,
    bf16_t* __restrict__ xb)
{
    size_t idx = (size_t)blockIdx.x * blockDim.x + threadIdx.x;
    if (idx < (size_t)M_TOTAL * D_MODEL) {
        int d = (int)(idx & (D_MODEL - 1));
        int n = (int)((idx >> 9) & (N_NODES - 1));
        xb[idx] = (bf16_t)(x[idx] + pos[(size_t)n * D_MODEL + d]);
    }
}

// ---------------------------------------------------------------------------
// 2/5) GEMM:  out[m, n] = sum_k A[m,k] * W[n,k] + bias[n]
//   A    : [4096, 512] bf16 row-major (tile DMA'd to LDS via TDM)
//   W    : [512, 512]  f32 row-major (torch Linear weight; row = out feature)
//   mode 0: write bf16 head-split out_hs[b, h, node, dh]  (qkv)
//   mode 1: write f32 flat out_f[m, n]                    (final projection)
// Block: 256 threads = 8 waves; tile M=128 (16 rows/wave), N=64, K-step 32.
// ---------------------------------------------------------------------------
__global__ __launch_bounds__(256) void gemm_kernel(
    const bf16_t* __restrict__ A, const float* __restrict__ W,
    const float* __restrict__ bias,
    bf16_t* __restrict__ out_hs, float* __restrict__ out_f, int mode)
{
    __shared__ bf16_t As[128][32];
    __shared__ bf16_t Bs[64][32];

    const int tid  = threadIdx.x;
    const int wave = tid >> 5;
    const int lane = tid & 31;
    const int half = lane >> 4;
    const int l16  = lane & 15;

    const int m0 = blockIdx.y * 128;
    const int n0 = blockIdx.x * 64;

    const unsigned as_lds = (unsigned)(uintptr_t)&As[0][0];

    v8f c[4] = {};

    for (int k0 = 0; k0 < D_MODEL; k0 += 32) {
        // --- stage A tile 128x32 via Tensor Data Mover (wave 0 only) ---
        if (wave == 0) {
            tdm_load_tile_bf16(A + (size_t)m0 * D_MODEL + k0, as_lds);
        }
        // --- stage B tile 64x32: Bs[n][k] = bf16(W[n0+n][k0+k]) ---
        {
            int row = tid >> 2;
            int col = (tid & 3) * 8;
            const float* src = W + (size_t)(n0 + row) * D_MODEL + k0 + col;
#pragma unroll
            for (int j = 0; j < 8; ++j) Bs[row][col + j] = (bf16_t)src[j];
        }
        if (wave == 0) {
            __builtin_amdgcn_s_wait_tensorcnt(0);
        }
        __syncthreads();

        // --- A fragment: row = wave*16 + l16 ; K halves per ISA layout ---
        const int arow = wave * 16 + l16;
        Frag a;
        a.h[0] = *(const v8bf*)&As[arow][half * 8];
        a.h[1] = *(const v8bf*)&As[arow][16 + half * 8];

        // batch all B fragments, then issue the 4 WMMAs back-to-back
        Frag bfr[4];
#pragma unroll
        for (int nt = 0; nt < 4; ++nt) {
            const int brow = nt * 16 + l16;
            bfr[nt].h[0] = *(const v8bf*)&Bs[brow][half * 8];
            bfr[nt].h[1] = *(const v8bf*)&Bs[brow][16 + half * 8];
        }
#pragma unroll
        for (int nt = 0; nt < 4; ++nt) {
            c[nt] = __builtin_amdgcn_wmma_f32_16x16x32_bf16(
                false, a.v, false, bfr[nt].v, (short)0, c[nt], false, false);
        }
        __syncthreads();
    }

    // --- epilogue ---
    const int rbase = m0 + wave * 16 + half * 8;
#pragma unroll
    for (int nt = 0; nt < 4; ++nt) {
        const int gcol = n0 + nt * 16 + l16;
        const float bn = bias[gcol];
#pragma unroll
        for (int i = 0; i < 8; ++i) {
            const int rg  = rbase + i;
            const float val = c[nt][i] + bn;
            if (mode == 0) {
                const int b  = rg >> 11;
                const int n  = rg & (N_NODES - 1);
                const int h  = gcol >> 6;
                const int dh = gcol & (HEAD_DIM - 1);
                out_hs[((size_t)(b * N_HEADS + h) * N_NODES + n) * HEAD_DIM + dh] =
                    (bf16_t)val;
            } else {
                out_f[(size_t)rg * D_MODEL + gcol] = val;
            }
        }
    }
}

// ---------------------------------------------------------------------------
// 3) vt[bh, dh, n] = v[bh, n, dh]
// ---------------------------------------------------------------------------
__global__ __launch_bounds__(256) void transpose_v_kernel(
    const bf16_t* __restrict__ v, bf16_t* __restrict__ vt)
{
    size_t idx = (size_t)blockIdx.x * blockDim.x + threadIdx.x;
    if (idx < (size_t)2 * N_HEADS * N_NODES * HEAD_DIM) {
        int dh = (int)(idx & (HEAD_DIM - 1));
        int n  = (int)((idx >> 6) & (N_NODES - 1));
        int bh = (int)(idx >> 17);
        vt[((size_t)bh * HEAD_DIM + dh) * N_NODES + n] = v[idx];
    }
}

// ---------------------------------------------------------------------------
// 4) Flash attention with topology bias + key mask.
//    One wave owns 16 query rows of one (b,h). 32-key chunks:
//      S   = (q @ k^T) * scale + prop . dist_enc   (4 WMMA + VALU bias)
//      online softmax (half-wave shfl_xor row reductions)
//      O  += P @ V                                 (4 WMMA, P via LDS restage)
// ---------------------------------------------------------------------------
__global__ __launch_bounds__(256) void attn_kernel(
    const bf16_t* __restrict__ q, const bf16_t* __restrict__ k,
    const bf16_t* __restrict__ vt, const float* __restrict__ prop,
    const float* __restrict__ dist_enc,
    const unsigned char* __restrict__ node_mask,
    bf16_t* __restrict__ attn_out)
{
    __shared__ bf16_t Ps[8][16][32];   // per-wave P staging (C-layout -> A-layout)

    const int tid  = threadIdx.x;
    const int wave = tid >> 5;
    const int lane = tid & 31;
    const int half = lane >> 4;
    const int l16  = lane & 15;

    const int gw    = blockIdx.x * 8 + wave;      // 0..2047
    const int qblk  = gw & 127;
    const int h     = (gw >> 7) & (N_HEADS - 1);
    const int b     = gw >> 10;
    const int qbase = qblk * 16;
    const int bh    = b * N_HEADS + h;

    // q A-fragments for K=64 (two 16x32 steps); lane row = qbase + l16
    const bf16_t* qp = q + ((size_t)bh * N_NODES + qbase + l16) * HEAD_DIM;
    Frag qf[2];
    qf[0].h[0] = *(const v8bf*)(qp + half * 8);
    qf[0].h[1] = *(const v8bf*)(qp + 16 + half * 8);
    qf[1].h[0] = *(const v8bf*)(qp + 32 + half * 8);
    qf[1].h[1] = *(const v8bf*)(qp + 48 + half * 8);

    float de[4];
#pragma unroll
    for (int j = 0; j < 4; ++j) de[j] = dist_enc[j * N_HEADS + h];

    float mrow[8], lrow[8];
#pragma unroll
    for (int i = 0; i < 8; ++i) { mrow[i] = -1e30f; lrow[i] = 0.0f; }
    v8f o[4] = {};

    const float scale = 0.125f;  // 1/sqrt(64)
    const bf16_t* kmat = k  + (size_t)bh * N_NODES * HEAD_DIM;
    const bf16_t* vtp  = vt + (size_t)bh * HEAD_DIM * N_NODES;

    for (int kc = 0; kc < N_NODES; kc += 32) {
        // prefetch next chunk's K rows / V^T rows (global_prefetch_b8)
        if (kc + 32 < N_NODES) {
            __builtin_prefetch(kmat + (size_t)(kc + 32 + lane) * HEAD_DIM, 0, 0);
            __builtin_prefetch(vtp + (size_t)lane * N_NODES + kc + 32, 0, 0);
        }

        // batch both 16-key subtiles' B fragments (K=64 -> 2 k-steps each)
        Frag kf[2][2];
#pragma unroll
        for (int sub = 0; sub < 2; ++sub) {
            const int key = kc + sub * 16 + l16;
            const bf16_t* kp = kmat + (size_t)key * HEAD_DIM;
            kf[sub][0].h[0] = *(const v8bf*)(kp + half * 8);
            kf[sub][0].h[1] = *(const v8bf*)(kp + 16 + half * 8);
            kf[sub][1].h[0] = *(const v8bf*)(kp + 32 + half * 8);
            kf[sub][1].h[1] = *(const v8bf*)(kp + 48 + half * 8);
        }

        v8f s[2];
#pragma unroll
        for (int sub = 0; sub < 2; ++sub) {
            v8f acc = {};
            acc = __builtin_amdgcn_wmma_f32_16x16x32_bf16(
                false, qf[0].v, false, kf[sub][0].v, (short)0, acc, false, false);
            acc = __builtin_amdgcn_wmma_f32_16x16x32_bf16(
                false, qf[1].v, false, kf[sub][1].v, (short)0, acc, false, false);

            // topology bias + key mask; this lane's 8 elems share column `key`
            const int key = kc + sub * 16 + l16;
            const bool msk = node_mask[(size_t)b * N_NODES + key] != 0;
#pragma unroll
            for (int i = 0; i < 8; ++i) {
                const int n = qbase + half * 8 + i;
                const float4 p4 =
                    *(const float4*)(prop + ((size_t)n * N_NODES + key) * 4);
                const float bias =
                    p4.x * de[0] + p4.y * de[1] + p4.z * de[2] + p4.w * de[3];
                acc[i] = msk ? (acc[i] * scale + bias) : -3.0e38f;
            }
            s[sub] = acc;
        }

        // online softmax update (per accumulator row; 16-lane half reductions)
#pragma unroll
        for (int i = 0; i < 8; ++i) {
            float t = fmaxf(s[0][i], s[1][i]);
            t = fmaxf(t, __shfl_xor(t, 1));
            t = fmaxf(t, __shfl_xor(t, 2));
            t = fmaxf(t, __shfl_xor(t, 4));
            t = fmaxf(t, __shfl_xor(t, 8));
            const float mnew  = fmaxf(mrow[i], t);
            const float alpha = __expf(mrow[i] - mnew);
            const float p0 = __expf(s[0][i] - mnew);
            const float p1 = __expf(s[1][i] - mnew);
            float rs = p0 + p1;
            rs += __shfl_xor(rs, 1);
            rs += __shfl_xor(rs, 2);
            rs += __shfl_xor(rs, 4);
            rs += __shfl_xor(rs, 8);
            lrow[i] = lrow[i] * alpha + rs;
            mrow[i] = mnew;
            s[0][i] = p0;
            s[1][i] = p1;
            o[0][i] *= alpha; o[1][i] *= alpha; o[2][i] *= alpha; o[3][i] *= alpha;
        }

        // restage P (16x32) through LDS: C-layout scatter -> A-layout gather
#pragma unroll
        for (int i = 0; i < 8; ++i) {
            const int row = half * 8 + i;
            Ps[wave][row][l16]      = (bf16_t)s[0][i];
            Ps[wave][row][16 + l16] = (bf16_t)s[1][i];
        }
        asm volatile("s_wait_dscnt 0" ::: "memory");
        Frag pf;
        pf.h[0] = *(const v8bf*)&Ps[wave][l16][half * 8];
        pf.h[1] = *(const v8bf*)&Ps[wave][l16][16 + half * 8];

        // O += P @ V : batch all 4 V^T fragments, then 4 WMMAs back-to-back
        Frag vf[4];
#pragma unroll
        for (int nt = 0; nt < 4; ++nt) {
            const int dh = nt * 16 + l16;
            const bf16_t* vp = vtp + (size_t)dh * N_NODES + kc + half * 8;
            vf[nt].h[0] = *(const v8bf*)(vp);
            vf[nt].h[1] = *(const v8bf*)(vp + 16);
        }
#pragma unroll
        for (int nt = 0; nt < 4; ++nt) {
            o[nt] = __builtin_amdgcn_wmma_f32_16x16x32_bf16(
                false, pf.v, false, vf[nt].v, (short)0, o[nt], false, false);
        }
        asm volatile("s_wait_dscnt 0" ::: "memory");
    }

    // epilogue: normalize, merge heads -> attn_out[b*N + n][h*64 + dh] (bf16)
#pragma unroll
    for (int nt = 0; nt < 4; ++nt) {
        const int col = h * HEAD_DIM + nt * 16 + l16;
#pragma unroll
        for (int i = 0; i < 8; ++i) {
            const int n = qbase + half * 8 + i;
            const float val = o[nt][i] / lrow[i];
            attn_out[((size_t)(b * N_NODES + n)) * D_MODEL + col] = (bf16_t)val;
        }
    }
}

// ---------------------------------------------------------------------------
// Launcher
// ---------------------------------------------------------------------------
extern "C" void kernel_launch(void* const* d_in, const int* in_sizes, int n_in,
                              void* d_out, int out_size, void* d_ws, size_t ws_size,
                              hipStream_t stream) {
    const float*         x         = (const float*)d_in[0];
    const unsigned char* node_mask = (const unsigned char*)d_in[1];
    const float*         prop      = (const float*)d_in[2];
    const float*         Wq        = (const float*)d_in[3];
    const float*         bq        = (const float*)d_in[4];
    const float*         Wk        = (const float*)d_in[5];
    const float*         bk        = (const float*)d_in[6];
    const float*         Wv        = (const float*)d_in[7];
    const float*         bv        = (const float*)d_in[8];
    const float*         Wo        = (const float*)d_in[9];
    const float*         bo        = (const float*)d_in[10];
    const float*         pos       = (const float*)d_in[11];
    const float*         dist      = (const float*)d_in[12];

    char* ws = (char*)d_ws;
    bf16_t* xb = (bf16_t*)(ws);
    bf16_t* qb = (bf16_t*)(ws + ((size_t)4  << 20));
    bf16_t* kb = (bf16_t*)(ws + ((size_t)8  << 20));
    bf16_t* vb = (bf16_t*)(ws + ((size_t)12 << 20));
    bf16_t* vt = (bf16_t*)(ws + ((size_t)16 << 20));
    bf16_t* ao = (bf16_t*)(ws + ((size_t)20 << 20));

    prep_x_kernel<<<(M_TOTAL * D_MODEL) / 256, 256, 0, stream>>>(x, pos, xb);

    dim3 ggrid(D_MODEL / 64, M_TOTAL / 128);  // (8, 32)
    gemm_kernel<<<ggrid, 256, 0, stream>>>(xb, Wq, bq, qb, nullptr, 0);
    gemm_kernel<<<ggrid, 256, 0, stream>>>(xb, Wk, bk, kb, nullptr, 0);
    gemm_kernel<<<ggrid, 256, 0, stream>>>(xb, Wv, bv, vb, nullptr, 0);

    transpose_v_kernel<<<(2 * N_HEADS * N_NODES * HEAD_DIM) / 256, 256, 0, stream>>>(vb, vt);

    attn_kernel<<<2048 / 8, 256, 0, stream>>>(qb, kb, vt, prop, dist, node_mask, ao);

    gemm_kernel<<<ggrid, 256, 0, stream>>>(ao, Wo, bo, nullptr, (float*)d_out, 1);
}
